// SetAbstraction_61280593379861
// MI455X (gfx1250) — matrix operand order repaired and checked by reading, not compiled, not run on hardware
//
#include <hip/hip_runtime.h>
#include <hip/hip_bf16.h>

// ---------------------------------------------------------------------------
// Set-abstraction block for MI455X (gfx1250, wave32, WMMA).
// All GEMM-like stages use V_WMMA_F32_16X16X32_F16 (f32 accumulate).
// Attention core exploits separability of conv2d(q - k) to avoid the
// 167MB (B,64,2048,160) intermediate entirely.
// ---------------------------------------------------------------------------

typedef __attribute__((ext_vector_type(16))) _Float16 v16h;
typedef __attribute__((ext_vector_type(8)))  float    v8f;

static constexpr int B_  = 2;
static constexpr int N_  = 8192;
static constexpr int M_  = 2048;   // N / STRIDE
static constexpr int C_  = 64;     // CIN
static constexpr int O_  = 128;    // COUT
static constexpr int L_  = 16;     // LPC
static constexpr int K_  = 160;    // top-k
static constexpr int S_  = 32;     // NSAMPLE
static constexpr float R2_ = 0.01f; // RADIUS^2

// ---------------------------------------------------------------------------
// WMMA fragment loaders (wave32), contiguous k order k = 16*(lane/16) + i, used for A and B alike:
//  A (16xK f16): lane L<16 -> row L, Ks 0..15;
//                lane L>=16 -> row L-16, Ks 16..31.
//  B (Kx16 f16): lanes 0-15 hold K 0..15 of column (lane), lanes 16-31 hold
//                K 16..31 of column (lane-16).
//  C/D (16x16 f32): lane n (&15) = column, vgpr v -> row (lane>>4)*8 + v.
// ---------------------------------------------------------------------------
__device__ __forceinline__ v16h load_a_frag(const _Float16* __restrict__ A,
                                            int lda, int rowBase, int kBase, int kMax) {
  const int lane = threadIdx.x & 31;
  const int r  = rowBase + (lane & 15);
  const int kb = (lane >> 4) ? 16 : 0;
  v16h a;
#pragma unroll
  for (int i = 0; i < 16; ++i) {
    int k0 = kBase + kb + i;
    a[i] = (k0 < kMax) ? A[r * lda + k0] : (_Float16)0.f;
  }
  return a;
}

__device__ __forceinline__ v16h load_b_frag(const _Float16* __restrict__ X,
                                            int ldx, int kBase, int colBase, int kMax) {
  const int lane = threadIdx.x & 31;
  const int n  = colBase + (lane & 15);
  const int kb = kBase + ((lane >> 4) ? 16 : 0);
  v16h b;
#pragma unroll
  for (int i = 0; i < 16; ++i) {
    int k = kb + i;
    b[i] = (k < kMax) ? X[k * ldx + n] : (_Float16)0.f;
  }
  return b;
}

__device__ __forceinline__ v8f wmma_f16(v16h a, v16h b, v8f c) {
  return __builtin_amdgcn_wmma_f32_16x16x32_f16(false, a, false, b, (short)0, c,
                                                false, false);
}

// ---------------------------------------------------------------------------
// small helpers
// ---------------------------------------------------------------------------
__global__ void cvt_f16_kernel(const float* __restrict__ src,
                               _Float16* __restrict__ dst, int n) {
  int i = blockIdx.x * blockDim.x + threadIdx.x;
  if (i < n) dst[i] = (_Float16)src[i];
}

// bn params (4,c): g, beta, mean, var  ->  s = g*rsqrt(var+eps), t = beta - mean*s
__global__ void bn_st_kernel(const float* __restrict__ prm,
                             float* __restrict__ s, float* __restrict__ t, int c) {
  int i = blockIdx.x * blockDim.x + threadIdx.x;
  if (i < c) {
    float g = prm[i], be = prm[c + i], mu = prm[2 * c + i], va = prm[3 * c + i];
    float sv = g * rsqrtf(va + 1e-5f);
    s[i] = sv;
    t[i] = be - mu * sv;
  }
}

// ---------------------------------------------------------------------------
// 1) Farthest point sampling: one block per batch, points resident in VGPRs.
// ---------------------------------------------------------------------------
__global__ void __launch_bounds__(1024) fps_kernel(const float* __restrict__ p,
                                                   int* __restrict__ idx) {
  const int b = blockIdx.x, tid = threadIdx.x;
  float px[8], py[8], pz[8], dist[8];
#pragma unroll
  for (int j = 0; j < 8; ++j) {
    const float* pp = p + ((long long)b * N_ + tid * 8 + j) * 3;
    px[j] = pp[0]; py[j] = pp[1]; pz[j] = pp[2];
    dist[j] = 1e10f;
  }
  __shared__ float redV[32];
  __shared__ int   redI[32];
  __shared__ float lastp[3];
  __shared__ int   bestS;
  if (tid == 0) {
    idx[b * M_] = 0;
    lastp[0] = p[(long long)b * N_ * 3 + 0];
    lastp[1] = p[(long long)b * N_ * 3 + 1];
    lastp[2] = p[(long long)b * N_ * 3 + 2];
  }
  __syncthreads();
  const int lane = tid & 31, warp = tid >> 5;
  for (int step = 1; step < M_; ++step) {
    float lx = lastp[0], ly = lastp[1], lz = lastp[2];
    float bv = -1.f; int bi = 0;
#pragma unroll
    for (int j = 0; j < 8; ++j) {
      float dx = px[j] - lx, dy = py[j] - ly, dz = pz[j] - lz;
      float d = fminf(dist[j], dx * dx + dy * dy + dz * dz);
      dist[j] = d;
      int i = tid * 8 + j;
      if (d > bv) { bv = d; bi = i; }   // j ascending => min index kept on tie
    }
    for (int off = 16; off > 0; off >>= 1) {
      float ov = __shfl_down(bv, off, 32);
      int   oi = __shfl_down(bi, off, 32);
      if (ov > bv || (ov == bv && oi < bi)) { bv = ov; bi = oi; }
    }
    if (lane == 0) { redV[warp] = bv; redI[warp] = bi; }
    __syncthreads();
    if (warp == 0) {
      bv = redV[lane]; bi = redI[lane];
      for (int off = 16; off > 0; off >>= 1) {
        float ov = __shfl_down(bv, off, 32);
        int   oi = __shfl_down(bi, off, 32);
        if (ov > bv || (ov == bv && oi < bi)) { bv = ov; bi = oi; }
      }
      if (lane == 0) { idx[b * M_ + step] = bi; bestS = bi; }
    }
    __syncthreads();
    int gbi = bestS;
    if (tid == (gbi >> 3)) {
      int j = gbi & 7;
      lastp[0] = px[j]; lastp[1] = py[j]; lastp[2] = pz[j];
    }
    __syncthreads();
  }
}

// ---------------------------------------------------------------------------
// 2) Gather sampled points/features + sigmoid gate, emit f16 copies.
// ---------------------------------------------------------------------------
__global__ void __launch_bounds__(256) gather_att_kernel(
    const float* __restrict__ p, const float* __restrict__ lp,
    const float* __restrict__ f, const float* __restrict__ Wg,
    const float* __restrict__ bg, const int* __restrict__ idx,
    float* __restrict__ newp_out, float* __restrict__ lpi_out,
    float* __restrict__ lpiw, float* __restrict__ fi,
    _Float16* __restrict__ fih, _Float16* __restrict__ fa16,
    float* __restrict__ attb) {
  const int b = blockIdx.y;
  const int j = blockIdx.x * 256 + threadIdx.x;
  const int id = idx[b * M_ + j];
  const float* pb = p + ((long long)b * N_ + id) * 3;
  float* np = newp_out + ((long long)b * M_ + j) * 3;
  np[0] = pb[0]; np[1] = pb[1]; np[2] = pb[2];
  float fv[C_];
  float acc = bg[0];
#pragma unroll
  for (int c = 0; c < C_; ++c) {
    float v = f[((long long)b * C_ + c) * N_ + id];
    fv[c] = v;
    acc += Wg[c] * v;
    long long o = ((long long)b * C_ + c) * M_ + j;
    fi[o] = v; fih[o] = (_Float16)v;
  }
  float a = 1.f / (1.f + expf(-acc));
  attb[b * M_ + j] = a;
#pragma unroll
  for (int c = 0; c < C_; ++c)
    fa16[((long long)b * C_ + c) * M_ + j] = (_Float16)(fv[c] * a);
#pragma unroll
  for (int c = 0; c < L_; ++c) {
    float l = lp[((long long)b * L_ + c) * N_ + id];
    long long o = ((long long)b * L_ + c) * M_ + j;
    lpi_out[o] = l; lpiw[o] = l;
  }
}

// ---------------------------------------------------------------------------
// 3) Generic WMMA GEMM: Y = act(s[o] * (A(OCxIC) @ X(ICxNC)) + t[o]).
//    One wave per 16x16 output tile.
// ---------------------------------------------------------------------------
__global__ void __launch_bounds__(32) wmma_gemm_kernel(
    const _Float16* __restrict__ A, const _Float16* __restrict__ X,
    float* __restrict__ Y, const float* __restrict__ svec,
    const float* __restrict__ tvec, int IC, int NC, int relu,
    long long xbs, long long ybs) {
  const int lane = threadIdx.x & 31;
  const int colBase = blockIdx.x * 16;
  const int rowBase = blockIdx.y * 16;
  const int b = blockIdx.z;
  const _Float16* Xb = X + (long long)b * xbs;
  float* Yb = Y + (long long)b * ybs;
  v8f acc = {0.f, 0.f, 0.f, 0.f, 0.f, 0.f, 0.f, 0.f};
  const int ktiles = (IC + 31) / 32;
  for (int kt = 0; kt < ktiles; ++kt) {
    if (kt + 1 < ktiles)
      __builtin_prefetch(Xb + (long long)(kt * 32 + 32) * NC + colBase, 0, 1);
    v16h a  = load_a_frag(A, IC, rowBase, kt * 32, IC);
    v16h bm = load_b_frag(Xb, NC, kt * 32, colBase, IC);
    acc = wmma_f16(a, bm, acc);
  }
  const int n  = colBase + (lane & 15);
  const int mb = (lane >> 4) * 8;
#pragma unroll
  for (int v = 0; v < 8; ++v) {
    int o = rowBase + mb + v;
    float y = acc[v];
    if (svec) y *= svec[o];
    if (tvec) y += tvec[o];
    if (relu) y = fmaxf(y, 0.f);
    Yb[(long long)o * NC + n] = y;
  }
}

// ---------------------------------------------------------------------------
// 4) Top-160 by iterative block argmax (mean over k is permutation invariant,
//    only the index set matters; tie-break = min index like lax.top_k).
// ---------------------------------------------------------------------------
__global__ void __launch_bounds__(256) topk_kernel(const float* __restrict__ attb,
                                                   int* __restrict__ topk) {
  __shared__ float sv[M_];
  __shared__ float rv[256];
  __shared__ int   ri[256];
  const int b = blockIdx.x, tid = threadIdx.x;
  for (int i = tid; i < M_; i += 256) sv[i] = attb[b * M_ + i];
  __syncthreads();
  for (int t = 0; t < K_; ++t) {
    float bv = -1e30f; int bi = 0;
    for (int i = tid; i < M_; i += 256) {
      float v = sv[i];
      if (v > bv) { bv = v; bi = i; }
    }
    rv[tid] = bv; ri[tid] = bi;
    __syncthreads();
    for (int s = 128; s > 0; s >>= 1) {
      if (tid < s) {
        float ov = rv[tid + s]; int oi = ri[tid + s];
        if (ov > rv[tid] || (ov == rv[tid] && oi < ri[tid])) {
          rv[tid] = ov; ri[tid] = oi;
        }
      }
      __syncthreads();
    }
    if (tid == 0) { topk[b * K_ + t] = ri[0]; sv[ri[0]] = -1e30f; }
    __syncthreads();
  }
}

__global__ void sparse_gather_kernel(const float* __restrict__ fi,
                                     const int* __restrict__ topk,
                                     _Float16* __restrict__ sfh) {
  int b = blockIdx.x, k = threadIdx.x;  // blockDim = 160
  int id = topk[b * K_ + k];
  for (int c = 0; c < C_; ++c)
    sfh[((long long)b * C_ + c) * K_ + k] =
        (_Float16)fi[((long long)b * C_ + c) * M_ + id];
}

// ---------------------------------------------------------------------------
// 5) Fused attention core.
//    r1[o,m,k] = relu(A1'[o,m] - B1''[o,k])  (separable conv + folded BN)
//    y2 = W2 @ r1 + b; attfeat = mean_k relu(y2)*v; emit (attfeat - fi) f16.
//    16 waves/block, one query m per wave, 40 WMMAs per wave.
// ---------------------------------------------------------------------------
__global__ void __launch_bounds__(512) attention_kernel(
    const _Float16* __restrict__ Whw2,  // 64x16
    const float* __restrict__ A1p,      // B x 16 x M
    const float* __restrict__ B1s,      // B x 16 x K
    const float* __restrict__ vbuf,     // B x 64 x K
    const float* __restrict__ bw2,      // 64
    const float* __restrict__ fi,       // B x 64 x M
    _Float16* __restrict__ t16) {       // B x 64 x M
  __shared__ float vsS[C_ * K_];
  __shared__ float b1S[16 * K_];
  __shared__ float bwS[C_];
  const int b = blockIdx.y, tid = threadIdx.x;
  for (int i = tid; i < C_ * K_; i += 512) vsS[i] = vbuf[b * C_ * K_ + i];
  for (int i = tid; i < 16 * K_; i += 512) b1S[i] = B1s[b * 16 * K_ + i];
  if (tid < C_) bwS[tid] = bw2[tid];
  __syncthreads();

  const int wid = tid >> 5, lane = tid & 31;
  const int m = blockIdx.x * 16 + wid;
  const int nn = lane & 15, mb = (lane >> 4) * 8;

  v16h aw[4];
#pragma unroll
  for (int ct = 0; ct < 4; ++ct) aw[ct] = load_a_frag(Whw2, 16, ct * 16, 0, 16);
  float a1v[16];
#pragma unroll
  for (int o = 0; o < 16; ++o) a1v[o] = A1p[(b * 16 + o) * M_ + m];

  v8f zero = {0.f, 0.f, 0.f, 0.f, 0.f, 0.f, 0.f, 0.f};
  v8f acc[4];
#pragma unroll
  for (int ct = 0; ct < 4; ++ct) acc[ct] = zero;

  for (int kt = 0; kt < K_ / 16; ++kt) {
    v16h bf;
    if (lane < 16) {
      int k = kt * 16 + lane;
#pragma unroll
      for (int i = 0; i < 16; ++i)
        bf[i] = (_Float16)fmaxf(a1v[i] - b1S[i * K_ + k], 0.f);
    } else {
#pragma unroll
      for (int i = 0; i < 16; ++i) bf[i] = (_Float16)0.f;
    }
    int kk = kt * 16 + nn;
#pragma unroll
    for (int ct = 0; ct < 4; ++ct) {
      v8f d = wmma_f16(aw[ct], bf, zero);
#pragma unroll
      for (int v = 0; v < 8; ++v) {
        int c = ct * 16 + mb + v;
        acc[ct][v] += fmaxf(d[v] + bwS[c], 0.f) * vsS[c * K_ + kk];
      }
    }
  }
  // sum over the 16 k-columns held in each lane half
#pragma unroll
  for (int ct = 0; ct < 4; ++ct)
#pragma unroll
    for (int v = 0; v < 8; ++v) {
      float x = acc[ct][v];
      x += __shfl_xor(x, 1, 32);
      x += __shfl_xor(x, 2, 32);
      x += __shfl_xor(x, 4, 32);
      x += __shfl_xor(x, 8, 32);
      acc[ct][v] = x;
    }
  if (nn == 0) {
#pragma unroll
    for (int ct = 0; ct < 4; ++ct)
#pragma unroll
      for (int v = 0; v < 8; ++v) {
        int c = ct * 16 + mb + v;
        long long o = ((long long)b * C_ + c) * M_ + m;
        t16[o] = (_Float16)(acc[ct][v] * (1.f / K_) - fi[o]);
      }
  }
}

// ---------------------------------------------------------------------------
// 6) Ball query: one wave per query; ballot + prefix preserves ascending
//    candidate order (matches sort-then-truncate); pad with first hit.
// ---------------------------------------------------------------------------
__global__ void __launch_bounds__(256) ball_query_kernel(
    const float* __restrict__ newp, const float* __restrict__ p,
    int* __restrict__ gidx) {
  const int tid = threadIdx.x, lane = tid & 31, wid = tid >> 5;
  const int b = blockIdx.y;
  const int m = blockIdx.x * 8 + wid;
  const float* q = newp + ((long long)b * M_ + m) * 3;
  float qx = q[0], qy = q[1], qz = q[2];
  int* row = gidx + ((long long)b * M_ + m) * S_;
  int count = 0;
  for (int base = 0; base < N_; base += 32) {
    int i = base + lane;
    const float* pp = p + ((long long)b * N_ + i) * 3;
    float dx = pp[0] - qx, dy = pp[1] - qy, dz = pp[2] - qz;
    bool in = (dx * dx + dy * dy + dz * dz) < R2_;
    unsigned msk = (unsigned)__ballot(in);
    if (in) {
      int rank = count + __popc(msk & ((1u << lane) - 1u));
      if (rank < S_) row[rank] = i;
    }
    count += __popc(msk);
    if (count >= S_) break;
  }
  int f0 = row[0];  // query is itself a point -> count >= 1 always
  for (int s = count + lane; s < S_; s += 32) row[s] = f0;
}

// ---------------------------------------------------------------------------
// 7) Final fused kernel: per-wave per-query
//    dlpj -> WMMA loc1(16->32) -> LDS restage -> WMMA loc2(32->128)
//    -> relu + fj + max over 32 neighbors -> + global_f -> relu(+identity)
// ---------------------------------------------------------------------------
__global__ void __launch_bounds__(256) loc_final_kernel(
    const _Float16* __restrict__ WhL1,  // 32x16
    const _Float16* __restrict__ WhL2,  // 128x32
    const float* __restrict__ s1, const float* __restrict__ t1,
    const float* __restrict__ s2, const float* __restrict__ t2,
    const float* __restrict__ lp,     // B x 16 x N
    const float* __restrict__ lpiw,   // B x 16 x M
    const float* __restrict__ fpost,  // B x 128 x N
    const int* __restrict__ gidx,     // B x M x 32
    const float* __restrict__ gf,     // B x 128 x M
    const float* __restrict__ ident,  // B x 128 x M
    float* __restrict__ fout) {       // B x 128 x M
  __shared__ _Float16 stage[8][32 * 33];
  const int tid = threadIdx.x, wid = tid >> 5, lane = tid & 31;
  const int b = blockIdx.y;
  const int m = blockIdx.x * 8 + wid;
  const int* gi = gidx + ((long long)b * M_ + m) * S_;
  const int nn = lane & 15, mb = (lane >> 4) * 8;

  v16h aL1[2], aL2[8];
#pragma unroll
  for (int rt = 0; rt < 2; ++rt) aL1[rt] = load_a_frag(WhL1, 16, rt * 16, 0, 16);
#pragma unroll
  for (int rt = 0; rt < 8; ++rt) aL2[rt] = load_a_frag(WhL2, 32, rt * 16, 0, 32);

  v8f zero = {0.f, 0.f, 0.f, 0.f, 0.f, 0.f, 0.f, 0.f};

#pragma unroll
  for (int ct = 0; ct < 2; ++ct) {
    v16h bf;
    if (lane < 16) {
      int id = gi[ct * 16 + lane];
#pragma unroll
      for (int i = 0; i < 16; ++i)
        bf[i] = (_Float16)(lp[((long long)b * L_ + i) * N_ + id] -
                           lpiw[((long long)b * L_ + i) * M_ + m]);
    } else {
#pragma unroll
      for (int i = 0; i < 16; ++i) bf[i] = (_Float16)0.f;
    }
#pragma unroll
    for (int rt = 0; rt < 2; ++rt) {
      v8f d = wmma_f16(aL1[rt], bf, zero);
#pragma unroll
      for (int v = 0; v < 8; ++v) {
        int o = rt * 16 + mb + v;
        stage[wid][o * 33 + ct * 16 + nn] =
            (_Float16)fmaxf(d[v] * s1[o] + t1[o], 0.f);
      }
    }
  }
  __syncthreads();

  v16h b2[2];
#pragma unroll
  for (int ct = 0; ct < 2; ++ct) b2[ct] = load_b_frag(&stage[wid][0], 33, 0, ct * 16, 32);
  int id0 = gi[nn];
  int id1 = gi[16 + nn];
#pragma unroll
  for (int rt = 0; rt < 8; ++rt) {
    v8f d0 = wmma_f16(aL2[rt], b2[0], zero);
    v8f d1 = wmma_f16(aL2[rt], b2[1], zero);
#pragma unroll
    for (int v = 0; v < 8; ++v) {
      int c = rt * 16 + mb + v;
      float l0 = fmaxf(d0[v] * s2[c] + t2[c], 0.f);
      float l1 = fmaxf(d1[v] * s2[c] + t2[c], 0.f);
      float mx = fmaxf(fpost[((long long)b * O_ + c) * N_ + id0] + l0,
                       fpost[((long long)b * O_ + c) * N_ + id1] + l1);
      mx = fmaxf(mx, __shfl_xor(mx, 1, 32));
      mx = fmaxf(mx, __shfl_xor(mx, 2, 32));
      mx = fmaxf(mx, __shfl_xor(mx, 4, 32));
      mx = fmaxf(mx, __shfl_xor(mx, 8, 32));
      if (nn == 0) {
        long long oi = ((long long)b * O_ + c) * M_ + m;
        fout[oi] = fmaxf(mx + gf[oi] + ident[oi], 0.f);
      }
    }
  }
}

// ---------------------------------------------------------------------------
// host launch
// ---------------------------------------------------------------------------
extern "C" void kernel_launch(void* const* d_in, const int* in_sizes, int n_in,
                              void* d_out, int out_size, void* d_ws, size_t ws_size,
                              hipStream_t stream) {
  (void)in_sizes; (void)n_in; (void)out_size; (void)ws_size;
  const float* p      = (const float*)d_in[0];
  const float* lp     = (const float*)d_in[1];
  const float* f      = (const float*)d_in[2];
  const float* W_skip = (const float*)d_in[3];
  const float* b_skip = (const float*)d_in[4];
  const float* W_post = (const float*)d_in[5];
  const float* bn_post= (const float*)d_in[6];
  const float* W_loc1 = (const float*)d_in[7];
  const float* bn_loc1= (const float*)d_in[8];
  const float* W_loc2 = (const float*)d_in[9];
  const float* bn_loc2= (const float*)d_in[10];
  const float* W_g    = (const float*)d_in[11];
  const float* b_g    = (const float*)d_in[12];
  const float* W_v    = (const float*)d_in[13];
  const float* b_v    = (const float*)d_in[14];
  const float* W_w1   = (const float*)d_in[15];
  const float* bn_w1  = (const float*)d_in[16];
  const float* W_w2   = (const float*)d_in[17];
  const float* b_w2   = (const float*)d_in[18];
  const float* W_m    = (const float*)d_in[19];
  const float* bn_m   = (const float*)d_in[20];

  float* out      = (float*)d_out;
  float* out_newp = out;                              // B*M*3
  float* out_lpi  = out + (size_t)B_ * M_ * 3;        // B*16*M
  float* out_fout = out_lpi + (size_t)B_ * L_ * M_;   // B*128*M

  char* ws = (char*)d_ws;
  size_t off = 0;
  auto alloc = [&](size_t bytes) -> void* {
    off = (off + 255) & ~(size_t)255;
    void* ptr = ws + off;
    off += bytes;
    return ptr;
  };
  // ~19 MB of scratch total
  int*      idx    = (int*)     alloc((size_t)B_ * M_ * 4);
  float*    fi     = (float*)   alloc((size_t)B_ * C_ * M_ * 4);
  _Float16* fih    = (_Float16*)alloc((size_t)B_ * C_ * M_ * 2);
  _Float16* fa16   = (_Float16*)alloc((size_t)B_ * C_ * M_ * 2);
  float*    attb   = (float*)   alloc((size_t)B_ * M_ * 4);
  float*    ident  = (float*)   alloc((size_t)B_ * O_ * M_ * 4);
  int*      topk   = (int*)     alloc((size_t)B_ * K_ * 4);
  _Float16* sfh    = (_Float16*)alloc((size_t)B_ * C_ * K_ * 2);
  float*    vbuf   = (float*)   alloc((size_t)B_ * C_ * K_ * 4);
  float*    A1p    = (float*)   alloc((size_t)B_ * 16 * M_ * 4);
  float*    B1s    = (float*)   alloc((size_t)B_ * 16 * K_ * 4);
  _Float16* t16    = (_Float16*)alloc((size_t)B_ * C_ * M_ * 2);
  float*    gfb    = (float*)   alloc((size_t)B_ * O_ * M_ * 4);
  _Float16* fh     = (_Float16*)alloc((size_t)B_ * C_ * N_ * 2);
  float*    fpost  = (float*)   alloc((size_t)B_ * O_ * N_ * 4);
  int*      gidx   = (int*)     alloc((size_t)B_ * M_ * S_ * 4);
  float*    lpiw   = (float*)   alloc((size_t)B_ * L_ * M_ * 4);
  _Float16* WhSkip = (_Float16*)alloc(O_ * C_ * 2);
  _Float16* WhPost = (_Float16*)alloc(O_ * C_ * 2);
  _Float16* WhL1   = (_Float16*)alloc(32 * 16 * 2);
  _Float16* WhL2   = (_Float16*)alloc(O_ * 32 * 2);
  _Float16* WhV    = (_Float16*)alloc(C_ * C_ * 2);
  _Float16* WhW1   = (_Float16*)alloc(16 * C_ * 2);
  _Float16* WhW2   = (_Float16*)alloc(C_ * 16 * 2);
  _Float16* WhM    = (_Float16*)alloc(O_ * C_ * 2);
  float* s_post = (float*)alloc(O_ * 4); float* t_post = (float*)alloc(O_ * 4);
  float* s_l1   = (float*)alloc(32 * 4); float* t_l1   = (float*)alloc(32 * 4);
  float* s_l2   = (float*)alloc(O_ * 4); float* t_l2   = (float*)alloc(O_ * 4);
  float* s_w1   = (float*)alloc(16 * 4); float* t_w1   = (float*)alloc(16 * 4);
  float* s_m    = (float*)alloc(O_ * 4); float* t_m    = (float*)alloc(O_ * 4);

  // weight conversions + folded bn coefficients
  auto cvt = [&](const float* s, _Float16* d, int n) {
    cvt_f16_kernel<<<(n + 255) / 256, 256, 0, stream>>>(s, d, n);
  };
  cvt(W_skip, WhSkip, O_ * C_);
  cvt(W_post, WhPost, O_ * C_);
  cvt(W_loc1, WhL1, 32 * 16);
  cvt(W_loc2, WhL2, O_ * 32);
  cvt(W_v,    WhV,  C_ * C_);
  cvt(W_w1,   WhW1, 16 * C_);
  cvt(W_w2,   WhW2, C_ * 16);
  cvt(W_m,    WhM,  O_ * C_);
  cvt(f, fh, B_ * C_ * N_);
  bn_st_kernel<<<1, 128, 0, stream>>>(bn_post, s_post, t_post, O_);
  bn_st_kernel<<<1, 32,  0, stream>>>(bn_loc1, s_l1, t_l1, 32);
  bn_st_kernel<<<1, 128, 0, stream>>>(bn_loc2, s_l2, t_l2, O_);
  bn_st_kernel<<<1, 16,  0, stream>>>(bn_w1, s_w1, t_w1, 16);
  bn_st_kernel<<<1, 128, 0, stream>>>(bn_m, s_m, t_m, O_);

  // pipeline
  fps_kernel<<<B_, 1024, 0, stream>>>(p, idx);
  gather_att_kernel<<<dim3(M_ / 256, B_), 256, 0, stream>>>(
      p, lp, f, W_g, b_g, idx, out_newp, out_lpi, lpiw, fi, fih, fa16, attb);

  // identity = W_skip @ fi + b_skip
  wmma_gemm_kernel<<<dim3(M_ / 16, O_ / 16, B_), 32, 0, stream>>>(
      WhSkip, fih, ident, nullptr, b_skip, C_, M_, 0,
      (long long)C_ * M_, (long long)O_ * M_);

  topk_kernel<<<B_, 256, 0, stream>>>(attb, topk);
  sparse_gather_kernel<<<B_, K_, 0, stream>>>(fi, topk, sfh);

  // v = W_v @ sparse_f + b_v
  wmma_gemm_kernel<<<dim3(K_ / 16, C_ / 16, B_), 32, 0, stream>>>(
      WhV, sfh, vbuf, nullptr, b_v, C_, K_, 0,
      (long long)C_ * K_, (long long)C_ * K_);
  // A1' = s_w1*(W_w1 @ (fi*att)) + t_w1
  wmma_gemm_kernel<<<dim3(M_ / 16, 1, B_), 32, 0, stream>>>(
      WhW1, fa16, A1p, s_w1, t_w1, C_, M_, 0,
      (long long)C_ * M_, (long long)16 * M_);
  // B1'' = s_w1*(W_w1 @ sparse_f)
  wmma_gemm_kernel<<<dim3(K_ / 16, 1, B_), 32, 0, stream>>>(
      WhW1, sfh, B1s, s_w1, nullptr, C_, K_, 0,
      (long long)C_ * K_, (long long)16 * K_);

  attention_kernel<<<dim3(M_ / 16, B_), 512, 0, stream>>>(
      WhW2, A1p, B1s, vbuf, b_w2, fi, t16);

  // global_f = relu(bn_m(W_m @ (attfeat - fi)))
  wmma_gemm_kernel<<<dim3(M_ / 16, O_ / 16, B_), 32, 0, stream>>>(
      WhM, t16, gfb, s_m, t_m, C_, M_, 1,
      (long long)C_ * M_, (long long)O_ * M_);
  // fpost = relu(bn_post(W_post @ f)) over full N
  wmma_gemm_kernel<<<dim3(N_ / 16, O_ / 16, B_), 32, 0, stream>>>(
      WhPost, fh, fpost, s_post, t_post, C_, N_, 1,
      (long long)C_ * N_, (long long)O_ * N_);

  ball_query_kernel<<<dim3(M_ / 8, B_), 256, 0, stream>>>(out_newp, p, gidx);

  loc_final_kernel<<<dim3(M_ / 8, B_), 256, 0, stream>>>(
      WhL1, WhL2, s_l1, t_l1, s_l2, t_l2, lp, lpiw, fpost, gidx, gfb, ident,
      out_fout);
}
